// Loss_21586505629955
// MI455X (gfx1250) — matrix operand order repaired and checked by reading, not compiled, not run on hardware
//
#include <hip/hip_runtime.h>
#include <math.h>

// ---------------------------------------------------------------------------
// Problem constants (from reference): frames/renders (B,T,C,H,W)=(4,24,4,240,320)
// ---------------------------------------------------------------------------
#define B_    4
#define T_    24
#define C_    4
#define H_    240
#define W_    320
#define HW    (H_ * W_)          // 76800
#define CHW   (C_ * H_ * W_)     // 307200
#define NS    (B_ * T_)          // 96 slices
#define NP    (B_ * (T_ - 1))    // 92 temporal pairs
#define NL    33                 // 2*pad+1 lags, pad = 16

// workspace layout (floats)
#define WS_PART 0                          // 96 x 4  supervised partials
#define WS_MEAN (WS_PART + NS * 4)         // 96      per-slice mean
#define WS_INV  (WS_MEAN + NS)             // 96      per-slice 1/(std+(std==0))
#define WS_CC   (WS_INV + NS)              // 92 x 33 x 33 correlation maps
#define WS_ZMAX (WS_CC + NP * NL * NL)     // 92      per-pair max
#define WS_TOT  (WS_ZMAX + NP)

// zncc kernel LDS geometry
#define BROW_STRIDE 328                    // halfs; 656B -> conflict-free slots
#define BROWS       153                    // 152 window rows + 1 zero row
#define BB_BYTES    (BROWS * BROW_STRIDE * 2)   // 100368
#define AROW_F      384                    // padded a-row floats per wave
#define SMEM_BYTES  (BB_BYTES + 8 * AROW_F * 4) // 112656

typedef _Float16 v16h __attribute__((ext_vector_type(16)));
typedef _Float16 v8h  __attribute__((ext_vector_type(8)));
typedef float    v8f  __attribute__((ext_vector_type(8)));

// ---------------------------------------------------------------------------
// Kernel 0: zero the cc accumulation buffer (atomics accumulate into it)
// ---------------------------------------------------------------------------
__global__ void zero_cc_kernel(float* __restrict__ cc, int n) {
    int i = blockIdx.x * blockDim.x + threadIdx.x;
    if (i < n) cc[i] = 0.0f;
}

// ---------------------------------------------------------------------------
// Kernel 1: per-(b,t) mean / inv-std of renders over (C,H,W)  (96 slices)
// ---------------------------------------------------------------------------
__global__ __launch_bounds__(1024) void stats_kernel(const float* __restrict__ renders,
                                                     float* __restrict__ meanOut,
                                                     float* __restrict__ invOut) {
    const int s = blockIdx.x;
    const float* p = renders + (size_t)s * CHW;
    float s1 = 0.0f, s2 = 0.0f;
    for (int i = threadIdx.x; i < CHW; i += 1024) {
        float v = p[i];
        s1 += v;
        s2 += v * v;
    }
    __shared__ float a1[1024], a2[1024];
    a1[threadIdx.x] = s1;
    a2[threadIdx.x] = s2;
    __syncthreads();
    for (int o = 512; o > 0; o >>= 1) {
        if (threadIdx.x < o) {
            a1[threadIdx.x] += a1[threadIdx.x + o];
            a2[threadIdx.x] += a2[threadIdx.x + o];
        }
        __syncthreads();
    }
    if (threadIdx.x == 0) {
        float mn  = a1[0] / (float)CHW;
        float var = a2[0] / (float)CHW - mn * mn;
        var = var > 0.0f ? var : 0.0f;
        float sd  = sqrtf(var);
        float den = sd + (sd == 0.0f ? 1.0f : 0.0f);   // std + (std==0)
        meanOut[s] = mn;
        invOut[s]  = 1.0f / den;
    }
}

// ---------------------------------------------------------------------------
// Kernel 2: supervised loss partials per (b,t): S_in, S_out, S_rgb, W_in
// ---------------------------------------------------------------------------
__global__ __launch_bounds__(1024) void supervised_kernel(const float* __restrict__ frames,
                                                          const float* __restrict__ renders,
                                                          float* __restrict__ part) {
    const int bt = blockIdx.x;                 // 0..95
    const size_t baseA = ((size_t)bt * C_ + 3) * HW;  // alpha plane
    const size_t base0 = ((size_t)bt * C_ + 0) * HW;

    float sIn = 0.0f, sOut = 0.0f, sRgb = 0.0f, wIn = 0.0f;
    for (int pix = threadIdx.x; pix < HW; pix += 1024) {
        float ga = frames[baseA + pix];
        float ra = renders[baseA + pix];
        float da = fabsf(ga - ra);
        float m  = (ga > 0.0f) ? 1.0f : 0.0f;
        sIn  += da * m;
        sOut += da * (1.0f - m);
        wIn  += m;
        float acc = 0.0f;
        #pragma unroll
        for (int c = 0; c < 3; ++c) {
            float fg = frames[base0 + (size_t)c * HW + pix];
            float rr = renders[base0 + (size_t)c * HW + pix];
            acc += fabsf(fg * ga - rr * ra);
        }
        sRgb += acc * m;
    }
    __shared__ float r0[1024], r1[1024], r2[1024], r3[1024];
    r0[threadIdx.x] = sIn; r1[threadIdx.x] = sOut;
    r2[threadIdx.x] = sRgb; r3[threadIdx.x] = wIn;
    __syncthreads();
    for (int o = 512; o > 0; o >>= 1) {
        if (threadIdx.x < o) {
            r0[threadIdx.x] += r0[threadIdx.x + o];
            r1[threadIdx.x] += r1[threadIdx.x + o];
            r2[threadIdx.x] += r2[threadIdx.x + o];
            r3[threadIdx.x] += r3[threadIdx.x + o];
        }
        __syncthreads();
    }
    if (threadIdx.x == 0) {
        float* q = part + bt * 4;
        q[0] = r0[0]; q[1] = r1[0]; q[2] = r2[0]; q[3] = r3[0];
    }
}

// ---------------------------------------------------------------------------
// Kernel 3: ZNCC correlation maps via WMMA (v_wmma_f32_16x16x32_f16)
//
// Per a-row (c,h_a): D[dx, j] = sum_k  Ahankel[dx,k] * Bdense[k,j]
//   Ahankel[dx,k] = aPad[k + dx]           (aPad[i] = a_row[i-16], zero pad)
//   Bdense[k,j]   = b_row(h_a - (j-16))[k] (zero row when out of window)
// accumulated over all (c,h_a) into cc[j][dx] per pair.
//
// Workgroup = (pair g, channel c, image half). 8 waves x 15 h_a rows each.
// b-plane window staged once in LDS (f16, stride 328 halfs -> no bank
// conflicts, 16B-aligned b128 fragment loads). A fragments built from a
// rolling register window: 2 new 8-float runs per k-chunk feed all 3 m-tiles.
// ---------------------------------------------------------------------------
__device__ inline v8h load_run(const float* __restrict__ P, int off) {
    v8h r;
    #pragma unroll
    for (int i = 0; i < 8; ++i) r[i] = (_Float16)P[off + i];
    return r;
}

__global__ __launch_bounds__(256, 1) void zncc_kernel(const float* __restrict__ renders,
                                                      const float* __restrict__ meanIn,
                                                      const float* __restrict__ invIn,
                                                      float* __restrict__ cc) {
    extern __shared__ char smem[];
    _Float16* bb = (_Float16*)smem;
    float* arows = (float*)(smem + BB_BYTES);

    const int bid  = blockIdx.x;       // 0..735
    const int g    = bid >> 3;         // pair
    const int rem  = bid & 7;
    const int c    = rem >> 1;         // channel
    const int half = rem & 1;          // image half (h_a in [h0, h0+120))
    const int b = g / (T_ - 1), t = g % (T_ - 1);
    const int sA = b * T_ + t, sB = sA + 1;

    const float mA = meanIn[sA], iA = invIn[sA];
    const float mB = meanIn[sB], iB = invIn[sB];
    const float* aPlane = renders + ((size_t)sA * C_ + c) * HW;
    const float* bPlane = renders + ((size_t)sB * C_ + c) * HW;
    const int h0 = half * 120;

    // ---- stage normalized b-plane window (f16) cooperatively ----
    for (int idx = threadIdx.x; idx < BROWS * W_; idx += 256) {
        int slot = idx / W_;
        int w    = idx - slot * W_;
        int hb   = h0 - 16 + slot;
        float v  = 0.0f;
        if (slot < 152 && hb >= 0 && hb < H_) v = (bPlane[hb * W_ + w] - mB) * iB;
        bb[slot * BROW_STRIDE + w] = (_Float16)v;
    }
    __syncthreads();

    const int lane = threadIdx.x & 31;
    const int wave = threadIdx.x >> 5;
    const int m    = lane & 15;        // A row-in-tile; B/D column-in-tile
    const int p    = lane >> 4;        // lane-half selector
    float* P = arows + wave * AROW_F;  // wave-private padded a-row (f32)

    v8f acc[3][3];
    const v8f vzero = {0.f, 0.f, 0.f, 0.f, 0.f, 0.f, 0.f, 0.f};
    #pragma unroll
    for (int i = 0; i < 3; ++i)
        #pragma unroll
        for (int j = 0; j < 3; ++j) acc[i][j] = vzero;

    const int haBeg = h0 + wave * 15;
    const int base  = m + 8 * p;       // per-lane run origin within P

    for (int ha = haBeg; ha < haBeg + 15; ++ha) {
        // stage padded normalized a-row as f32 (P[i] = a_row[i-16])
        for (int i = lane; i < AROW_F; i += 32) {
            int w = i - 16;
            float v = 0.0f;
            if (w >= 0 && w < W_) v = (aPlane[ha * W_ + w] - mA) * iA;
            P[i] = v;
        }

        // per-lane LDS half-offsets for the 3 B n-tiles (column j = nt*16 + m)
        int slotAddr[3];
        #pragma unroll
        for (int nt = 0; nt < 3; ++nt) {
            int j  = nt * 16 + m;
            int sl = (ha - h0) + 32 - j;       // window slot of row h_a-(j-16)
            if (sl < 0) sl = 152;              // zero row (discarded columns)
            slotAddr[nt] = sl * BROW_STRIDE + p * 8;
        }

        v8h r0 = load_run(P, base);
        v8h r1 = load_run(P, base + 16);
        v8h r2 = load_run(P, base + 32);
        v8h r3 = load_run(P, base + 48);

        for (int kb = 0; kb < W_; kb += 32) {
            // A fragments (ISA A16x32 layout: per-lane K runs 8p..8p+7, 16+8p..)
            v16h A0 = __builtin_shufflevector(r0, r1, 0,1,2,3,4,5,6,7,8,9,10,11,12,13,14,15);
            v16h A1 = __builtin_shufflevector(r1, r2, 0,1,2,3,4,5,6,7,8,9,10,11,12,13,14,15);
            v16h A2 = __builtin_shufflevector(r2, r3, 0,1,2,3,4,5,6,7,8,9,10,11,12,13,14,15);

            #pragma unroll
            for (int nt = 0; nt < 3; ++nt) {
                int ho = slotAddr[nt] + kb;    // 16B aligned
                v8h blo = *(const v8h*)(bb + ho);
                v8h bhi = *(const v8h*)(bb + ho + 16);
                v16h Bf = __builtin_shufflevector(blo, bhi,
                            0,1,2,3,4,5,6,7,8,9,10,11,12,13,14,15);
                acc[0][nt] = __builtin_amdgcn_wmma_f32_16x16x32_f16(
                    false, A0, false, Bf, (short)0, acc[0][nt], false, false);
                acc[1][nt] = __builtin_amdgcn_wmma_f32_16x16x32_f16(
                    false, A1, false, Bf, (short)0, acc[1][nt], false, false);
                acc[2][nt] = __builtin_amdgcn_wmma_f32_16x16x32_f16(
                    false, A2, false, Bf, (short)0, acc[2][nt], false, false);
            }
            // roll the run window: 2 new runs serve the next k-chunk
            if (kb < W_ - 32) {
                r0 = r2;
                r1 = r3;
                r2 = load_run(P, base + kb + 64);
                r3 = load_run(P, base + kb + 80);
            }
        }
    }

    // ---- flush: D layout: column n = lane%16, row M = vgpr + 8p ----
    float* ccg = cc + (size_t)g * (NL * NL);
    #pragma unroll
    for (int mt = 0; mt < 3; ++mt)
        #pragma unroll
        for (int nt = 0; nt < 3; ++nt)
            #pragma unroll
            for (int r = 0; r < 8; ++r) {
                int dx = mt * 16 + r + 8 * p;
                int j  = nt * 16 + m;
                if (dx < NL && j < NL)
                    atomicAdd(&ccg[j * NL + dx], acc[mt][nt][r]);
            }
}

// ---------------------------------------------------------------------------
// Kernel 4: per-pair max over the 33x33 map
// ---------------------------------------------------------------------------
__global__ __launch_bounds__(256) void maxcc_kernel(const float* __restrict__ cc,
                                                    float* __restrict__ zmax) {
    const int g = blockIdx.x;
    const float* p = cc + (size_t)g * (NL * NL);
    float mx = -3.4e38f;
    for (int i = threadIdx.x; i < NL * NL; i += 256) mx = fmaxf(mx, p[i]);
    __shared__ float red[256];
    red[threadIdx.x] = mx;
    __syncthreads();
    for (int o = 128; o > 0; o >>= 1) {
        if (threadIdx.x < o)
            red[threadIdx.x] = fmaxf(red[threadIdx.x], red[threadIdx.x + o]);
        __syncthreads();
    }
    if (threadIdx.x == 0) zmax[g] = red[0];
}

// ---------------------------------------------------------------------------
// Kernel 5: assemble outputs: out[b*2+0] = sup, out[b*2+1] = tc
// ---------------------------------------------------------------------------
__global__ void final_kernel(const float* __restrict__ part,
                             const float* __restrict__ zmax,
                             float* __restrict__ out) {
    int b = threadIdx.x;
    if (b < B_) {
        float sup = 0.0f;
        for (int t = 0; t < T_; ++t) {
            const float* q = part + (b * T_ + t) * 4;
            float sIn = q[0], sOut = q[1], sRgb = q[2], wI = q[3];
            float w1 = wI + (wI == 0.0f ? 1.0f : 0.0f);
            float wO = (float)HW - wI;
            float w2 = wO + (wO == 0.0f ? 1.0f : 0.0f);
            sup += (sIn / w1 + sOut / w2 + sRgb / w1) * (1.0f / 3.0f);
        }
        sup /= (float)T_;
        float zm = 0.0f;
        for (int t = 0; t < T_ - 1; ++t) zm += zmax[b * (T_ - 1) + t] * (1.0f / (float)CHW);
        zm /= (float)(T_ - 1);
        out[b * 2 + 0] = sup;                 // W_SUPERVISED = 1
        out[b * 2 + 1] = (1.0f - zm) * 0.5f;  // W_TEMPORAL = 1
    }
}

// ---------------------------------------------------------------------------
extern "C" void kernel_launch(void* const* d_in, const int* in_sizes, int n_in,
                              void* d_out, int out_size, void* d_ws, size_t ws_size,
                              hipStream_t stream) {
    (void)in_sizes; (void)n_in; (void)out_size; (void)ws_size;

    const float* frames  = (const float*)d_in[0];
    const float* renders = (const float*)d_in[1];
    float* ws   = (float*)d_ws;
    float* part = ws + WS_PART;
    float* mean = ws + WS_MEAN;
    float* inv  = ws + WS_INV;
    float* cc   = ws + WS_CC;
    float* zmax = ws + WS_ZMAX;
    float* out  = (float*)d_out;

    // allow > default dynamic LDS (idempotent; safe under graph capture)
    (void)hipFuncSetAttribute((const void*)zncc_kernel,
                              hipFuncAttributeMaxDynamicSharedMemorySize,
                              SMEM_BYTES);

    int ccN = NP * NL * NL;
    zero_cc_kernel<<<(ccN + 255) / 256, 256, 0, stream>>>(cc, ccN);
    stats_kernel<<<NS, 1024, 0, stream>>>(renders, mean, inv);
    supervised_kernel<<<NS, 1024, 0, stream>>>(frames, renders, part);
    zncc_kernel<<<NP * C_ * 2, 256, SMEM_BYTES, stream>>>(renders, mean, inv, cc);
    maxcc_kernel<<<NP, 256, 0, stream>>>(cc, zmax);
    final_kernel<<<1, 32, 0, stream>>>(part, zmax, out);
}
